// LSTM_autoreg_torchscript_74234214744409
// MI455X (gfx1250) — matrix operand order repaired and checked by reading, not compiled, not run on hardware
//
#include <hip/hip_runtime.h>
#include <cstdint>

#define DI __device__ __forceinline__

typedef __attribute__((ext_vector_type(16))) _Float16 v16h;
typedef __attribute__((ext_vector_type(8)))  _Float16 v8h;
typedef __attribute__((ext_vector_type(8)))  float    v8f;

namespace {

constexpr int  kB   = 16384;   // batch
constexpr int  kT   = 60;      // time steps (NLEV)
constexpr long kOut1 = (long)kB * kT * 4;   // out      [B,60,4]
constexpr long kOut2 = (long)kB * 3;        // out_sfc  [B,3]

// ---- kernel1 LDS layout (bytes) ----
constexpr int SM1_W1   = 0;          // 256*96 f16  = 49152
constexpr int SM1_BIAS = 49152;      // 256 f32     = 1024
constexpr int SM1_H    = 50176;      // 4 waves * 16*64 f16 = 8192
constexpr int SM1_SIZE = 58368;

// ---- kernel2 LDS layout (bytes) ----
constexpr int SM2_W2    = 0;         // 256*128 f16 = 65536
constexpr int SM2_BIAS  = 65536;     // 256 f32     = 1024
constexpr int SM2_WLAT  = 66560;     // 16*64 f16   = 2048
constexpr int SM2_BLAT  = 68608;     // 16 f32      = 64
constexpr int SM2_WOUT  = 68672;     // 4*16 f32    = 256
constexpr int SM2_BOUT  = 68928;     // 4 f32       = 16
constexpr int SM2_WSFCO = 68944;     // 3*64 f32    = 768
constexpr int SM2_BSFCO = 69712;     // 3 f32 (+pad)= 16
constexpr int SM2_H     = 69728;     // 4 waves * 2048 = 8192
constexpr int SM2_LAT   = 77920;     // 4 waves * 1024 = 4096
constexpr int SM2_X     = 82016;     // 4 waves * 2 * 2048 = 16384 (double buffer)
constexpr int SM2_SIZE  = 98400;

} // namespace

DI float sigf(float x) { return 1.0f / (1.0f + __expf(-x)); }
DI float tanh_fast(float x) {
  x = fminf(15.0f, fmaxf(-15.0f, x));
  float e = __expf(-2.0f * x);
  return (1.0f - e) / (1.0f + e);
}

DI v8f wmma_f16(v16h a, v16h b, v8f c) {
  return __builtin_amdgcn_wmma_f32_16x16x32_f16(false, a, false, b, (short)0, c,
                                                false, false);
}

// A-fragment (16x32 f16) from an LDS tile stored row-major [16][64] halves.
// Lane layout: m = lane%16; elements 0-15 hold k in [k0+16*half, +16)
// (contiguous, matching the B fragments).
DI v16h load_a_frag(const _Float16* base, int m, int k0, int hh) {
  const v8h lo = *(const v8h*)(base + m * 64 + k0 + 16 * hh);
  const v8h hi = *(const v8h*)(base + m * 64 + k0 + 16 * hh + 8);
  return __builtin_shufflevector(lo, hi, 0, 1, 2, 3, 4, 5, 6, 7, 8, 9, 10, 11,
                                 12, 13, 14, 15);
}

DI v8f splat8(float x) {
  v8f r;
#pragma unroll
  for (int i = 0; i < 8; ++i) r[i] = x;
  return r;
}

// ---- CDNA5 async LDS <-> global (ASYNCcnt-tracked) ----
DI void async_load_b128(unsigned lds, unsigned long long gaddr) {
  asm volatile("global_load_async_to_lds_b128 %0, %1, off" ::"v"(lds),
               "v"(gaddr)
               : "memory");
}
DI void async_store_b128(unsigned long long gaddr, unsigned lds) {
  asm volatile("global_store_async_from_lds_b128 %0, %1, off" ::"v"(gaddr),
               "v"(lds)
               : "memory");
}
DI void wait_async0() { asm volatile("s_wait_asynccnt 0" ::: "memory"); }
DI void wait_ds0() { asm volatile("s_wait_dscnt 0" ::: "memory"); }

// =====================================================================
// Kernel 1: rnn1 over reversed time; emits rnn1out to scratch as f16
// in layout [t][B][64].
// =====================================================================
__global__ __launch_bounds__(128) void k_rnn1(
    const float* __restrict__ inputs_main,  // [B,60,4]
    const float* __restrict__ inputs_aux,   // [B,3]
    const float* __restrict__ rnn1_mem,     // [B,60,16]
    const float* __restrict__ Wsfc1, const float* __restrict__ bsfc1,
    const float* __restrict__ Wsfc2, const float* __restrict__ bsfc2,
    const float* __restrict__ Wih,   // [256,20]
    const float* __restrict__ Whh,   // [256,64]
    const float* __restrict__ bih, const float* __restrict__ bhh,
    _Float16* __restrict__ ws_rnn1out)  // [60][B][64] f16
{
  extern __shared__ char smem[];
  _Float16* W1    = (_Float16*)(smem + SM1_W1);     // [256][96]
  float*    bias1 = (float*)(smem + SM1_BIAS);      // [256]
  const int tid  = threadIdx.x;
  const int lane = tid & 31;
  const int wv   = tid >> 5;
  const int n    = lane & 15;
  const int hh   = lane >> 4;
  _Float16* hbuf = (_Float16*)(smem + SM1_H) + wv * (16 * 64);
  const long b0  = ((long)blockIdx.x * 4 + wv) * 16;

  // ---- stage combined weights [Wih(pad 32) | Whh] as f16 ----
  for (int idx = tid; idx < 256 * 96; idx += 128) {
    int g = idx / 96, k = idx % 96;
    float w = (k < 20) ? Wih[g * 20 + k]
                       : ((k < 32) ? 0.0f : Whh[g * 64 + (k - 32)]);
    W1[idx] = (_Float16)w;
  }
  for (int g = tid; g < 256; g += 128) bias1[g] = bih[g] + bhh[g];
  __syncthreads();

  // per-lane gate biases, hoisted out of the time loop: bpre[gi][jj]
  float bpre[4][4];
#pragma unroll
  for (int gi = 0; gi < 4; ++gi)
#pragma unroll
    for (int jj = 0; jj < 4; ++jj)
      bpre[gi][jj] = bias1[(gi * 4 + jj) * 16 + n];

  // ---- h0 = tanh(aux@Wsfc1.T+b1) into hbuf (A-friendly row-major) ----
  for (int q = lane; q < 16 * 64; q += 32) {
    int m = q >> 6, hc = q & 63;
    long b = b0 + m;
    float a0 = inputs_aux[b * 3 + 0];
    float a1 = inputs_aux[b * 3 + 1];
    float a2 = inputs_aux[b * 3 + 2];
    float hx = tanh_fast(a0 * Wsfc1[hc * 3 + 0] + a1 * Wsfc1[hc * 3 + 1] +
                         a2 * Wsfc1[hc * 3 + 2] + bsfc1[hc]);
    hbuf[q] = (_Float16)hx;
  }
  // ---- c0 = tanh(aux@Wsfc2.T+b2) directly in D layout ----
  v8f creg[4];
#pragma unroll
  for (int jj = 0; jj < 4; ++jj) {
#pragma unroll
    for (int v = 0; v < 8; ++v) {
      int m = hh * 8 + v, hc = jj * 16 + n;
      long b = b0 + m;
      float a0 = inputs_aux[b * 3 + 0];
      float a1 = inputs_aux[b * 3 + 1];
      float a2 = inputs_aux[b * 3 + 2];
      creg[jj][v] = tanh_fast(a0 * Wsfc2[hc * 3 + 0] + a1 * Wsfc2[hc * 3 + 1] +
                              a2 * Wsfc2[hc * 3 + 2] + bsfc2[hc]);
    }
  }

  v16h ah0 = load_a_frag(hbuf, n, 0, hh);
  v16h ah1 = load_a_frag(hbuf, n, 32, hh);

  for (int s = 0; s < kT; ++s) {
    const int t1 = kT - 1 - s;  // flipped time index

    // previous step's async hbuf->global store must be drained before we
    // overwrite hbuf below
    wait_async0();

    // ---- gather x fragment via 4 float4 loads: row = [main(4)|mem(16)|pad] ----
    v16h ax;
    {
      const long b = b0 + n;
      const float4* m4 = (const float4*)(inputs_main + ((long)b * kT + t1) * 4);
      const float4* x4 = (const float4*)(rnn1_mem + ((long)b * kT + t1) * 16);
      // hh==0: elems0-15 <- k0..15 = [m4[0], x4[0], x4[1], x4[2]]
      // hh==1: elems0-3 <- k16..19 = x4[3];  elems4-15 <- 0
      float4 l0 = hh ? x4[3] : m4[0];
      float4 l1 = x4[0];
      float4 l2 = x4[1];
      float4 l3 = x4[2];
      ax[0] = (_Float16)l0.x; ax[1] = (_Float16)l0.y;
      ax[2] = (_Float16)l0.z; ax[3] = (_Float16)l0.w;
      if (hh) {
        for (int i = 4; i < 16; ++i) ax[i] = (_Float16)0.f;
      } else {
        ax[4] = (_Float16)l1.x; ax[5] = (_Float16)l1.y;
        ax[6] = (_Float16)l1.z; ax[7] = (_Float16)l1.w;
        ax[8] = (_Float16)l2.x;  ax[9] = (_Float16)l2.y;
        ax[10] = (_Float16)l2.z; ax[11] = (_Float16)l2.w;
        ax[12] = (_Float16)l3.x; ax[13] = (_Float16)l3.y;
        ax[14] = (_Float16)l3.z; ax[15] = (_Float16)l3.w;
      }
    }

#pragma unroll
    for (int jj = 0; jj < 4; ++jj) {
      v8f acc[4] = {};  // zero C; bias folded into activations below
#pragma unroll
      for (int kt = 0; kt < 3; ++kt) {
        v16h a = (kt == 0) ? ax : ((kt == 1) ? ah0 : ah1);
        // load all 4 gate B-fragments first so WMMAs can overlap the loads
        v16h bf0 = *(const v16h*)(W1 + ((0 * 4 + jj) * 16 + n) * 96 + kt * 32 + 16 * hh);
        v16h bf1 = *(const v16h*)(W1 + ((1 * 4 + jj) * 16 + n) * 96 + kt * 32 + 16 * hh);
        v16h bf2 = *(const v16h*)(W1 + ((2 * 4 + jj) * 16 + n) * 96 + kt * 32 + 16 * hh);
        v16h bf3 = *(const v16h*)(W1 + ((3 * 4 + jj) * 16 + n) * 96 + kt * 32 + 16 * hh);
        acc[0] = wmma_f16(a, bf0, acc[0]);
        acc[1] = wmma_f16(a, bf1, acc[1]);
        acc[2] = wmma_f16(a, bf2, acc[2]);
        acc[3] = wmma_f16(a, bf3, acc[3]);
      }
#pragma unroll
      for (int v = 0; v < 8; ++v) {
        float iv = sigf(acc[0][v] + bpre[0][jj]);
        float fv = sigf(acc[1][v] + bpre[1][jj]);
        float gv = tanh_fast(acc[2][v] + bpre[2][jj]);
        float ov = sigf(acc[3][v] + bpre[3][jj]);
        float c = fv * creg[jj][v] + iv * gv;
        creg[jj][v] = c;
        hbuf[(hh * 8 + v) * 64 + jj * 16 + n] = (_Float16)(ov * tanh_fast(c));
      }
    }

    // ---- stream the finished 2KB h tile to scratch asynchronously ----
    wait_ds0();  // hbuf stores visible before async engine reads LDS
    {
      unsigned ldsb = (unsigned)(uintptr_t)hbuf;
      unsigned long long gb =
          (unsigned long long)(uintptr_t)(ws_rnn1out +
                                          ((long)t1 * kB + b0) * 64);
#pragma unroll
      for (int u = 0; u < 4; ++u)
        async_store_b128(gb + (unsigned long long)(u * 512 + lane * 16),
                         ldsb + (unsigned)(u * 512 + lane * 16));
    }

    // reload h fragments for next step (concurrent LDS reads are fine)
    ah0 = load_a_frag(hbuf, n, 0, hh);
    ah1 = load_a_frag(hbuf, n, 32, hh);
  }
  // implicit wait-idle at s_endpgm drains remaining async stores
}

// =====================================================================
// Kernel 2: rnn2 forward time + latent/out/out_sfc heads.
// =====================================================================
__global__ __launch_bounds__(128) void k_rnn2(
    const float* __restrict__ inputs_aux,
    const float* __restrict__ Wtoa1, const float* __restrict__ btoa1,
    const float* __restrict__ Wtoa2, const float* __restrict__ btoa2,
    const float* __restrict__ Wih,   // [256,64]
    const float* __restrict__ Whh,   // [256,64]
    const float* __restrict__ bih, const float* __restrict__ bhh,
    const float* __restrict__ Wlat,  // [16,64]
    const float* __restrict__ blat,
    const float* __restrict__ Wout,  // [4,16]
    const float* __restrict__ bout,
    const float* __restrict__ Wsfco,  // [3,64]
    const float* __restrict__ bsfco,
    const _Float16* __restrict__ xs,  // rnn1out scratch [60][B][64] f16
    float* __restrict__ out1,         // [B,60,4]
    float* __restrict__ out2,         // [B,3]
    float* __restrict__ out3)         // [B,60,16]
{
  extern __shared__ char smem[];
  _Float16* W2     = (_Float16*)(smem + SM2_W2);     // [256][128]
  float*    bias2  = (float*)(smem + SM2_BIAS);      // [256]
  _Float16* WlatL  = (_Float16*)(smem + SM2_WLAT);   // [16][64]
  float*    blatL  = (float*)(smem + SM2_BLAT);
  float*    WoutL  = (float*)(smem + SM2_WOUT);
  float*    boutL  = (float*)(smem + SM2_BOUT);
  float*    WsfcoL = (float*)(smem + SM2_WSFCO);
  float*    bsfcoL = (float*)(smem + SM2_BSFCO);

  const int tid  = threadIdx.x;
  const int lane = tid & 31;
  const int wv   = tid >> 5;
  const int n    = lane & 15;
  const int hh   = lane >> 4;
  _Float16* hbuf   = (_Float16*)(smem + SM2_H) + wv * (16 * 64);
  float*    latb   = (float*)(smem + SM2_LAT) + wv * 256;
  char*     xstage = smem + SM2_X + wv * 4096;  // 2 x 2KB double buffer
  const long b0 = ((long)blockIdx.x * 4 + wv) * 16;

  // ---- stage combined [Wih | Whh] -> [256][128] f16 ----
  for (int idx = tid; idx < 256 * 128; idx += 128) {
    int g = idx >> 7, k = idx & 127;
    float w = (k < 64) ? Wih[g * 64 + k] : Whh[g * 64 + (k - 64)];
    W2[idx] = (_Float16)w;
  }
  for (int g = tid; g < 256; g += 128) bias2[g] = bih[g] + bhh[g];
  for (int idx = tid; idx < 16 * 64; idx += 128)
    WlatL[idx] = (_Float16)Wlat[idx];
  if (tid < 16) blatL[tid] = blat[tid];
  if (tid < 64) WoutL[tid] = Wout[tid];
  if (tid < 4) boutL[tid] = bout[tid];
  for (int idx = tid; idx < 192; idx += 128) WsfcoL[idx] = Wsfco[idx];
  if (tid < 3) bsfcoL[tid] = bsfco[tid];
  __syncthreads();

  float bpre[4][4];
#pragma unroll
  for (int gi = 0; gi < 4; ++gi)
#pragma unroll
    for (int jj = 0; jj < 4; ++jj)
      bpre[gi][jj] = bias2[(gi * 4 + jj) * 16 + n];
  const float blv = blatL[n];

  // ---- h0 = toa*Wtoa2+btoa2 ; c0 = toa*Wtoa1+btoa1 (no tanh) ----
  for (int q = lane; q < 16 * 64; q += 32) {
    int m = q >> 6, hc = q & 63;
    float toa = inputs_aux[(b0 + m) * 3 + 1];
    hbuf[q] = (_Float16)(toa * Wtoa2[hc] + btoa2[hc]);
  }
  v8f creg[4];
#pragma unroll
  for (int jj = 0; jj < 4; ++jj) {
#pragma unroll
    for (int v = 0; v < 8; ++v) {
      int m = hh * 8 + v, hc = jj * 16 + n;
      float toa = inputs_aux[(b0 + m) * 3 + 1];
      creg[jj][v] = toa * Wtoa1[hc] + btoa1[hc];
    }
  }

  v16h ah0 = load_a_frag(hbuf, n, 0, hh);
  v16h ah1 = load_a_frag(hbuf, n, 32, hh);

  // ---- prefetch x tile for t = 0 (async global -> LDS) ----
  {
    unsigned ldsb = (unsigned)(uintptr_t)xstage;
    unsigned long long gb =
        (unsigned long long)(uintptr_t)(xs + ((long)0 * kB + b0) * 64);
#pragma unroll
    for (int u = 0; u < 4; ++u)
      async_load_b128(ldsb + (unsigned)(u * 512 + lane * 16),
                      gb + (unsigned long long)(u * 512 + lane * 16));
  }

  for (int t = 0; t < kT; ++t) {
    wait_async0();
    const _Float16* xb = (const _Float16*)(xstage + (t & 1) * 2048);
    v16h ax0 = load_a_frag(xb, n, 0, hh);
    v16h ax1 = load_a_frag(xb, n, 32, hh);

    // prefetch next step's x tile into the other buffer
    if (t + 1 < kT) {
      unsigned ldsb = (unsigned)(uintptr_t)(xstage + ((t + 1) & 1) * 2048);
      unsigned long long gb =
          (unsigned long long)(uintptr_t)(xs + ((long)(t + 1) * kB + b0) * 64);
#pragma unroll
      for (int u = 0; u < 4; ++u)
        async_load_b128(ldsb + (unsigned)(u * 512 + lane * 16),
                        gb + (unsigned long long)(u * 512 + lane * 16));
    }

#pragma unroll
    for (int jj = 0; jj < 4; ++jj) {
      v8f acc[4] = {};
#pragma unroll
      for (int kt = 0; kt < 4; ++kt) {
        v16h a = (kt == 0) ? ax0 : ((kt == 1) ? ax1 : ((kt == 2) ? ah0 : ah1));
        v16h bf0 = *(const v16h*)(W2 + ((0 * 4 + jj) * 16 + n) * 128 + kt * 32 + 16 * hh);
        v16h bf1 = *(const v16h*)(W2 + ((1 * 4 + jj) * 16 + n) * 128 + kt * 32 + 16 * hh);
        v16h bf2 = *(const v16h*)(W2 + ((2 * 4 + jj) * 16 + n) * 128 + kt * 32 + 16 * hh);
        v16h bf3 = *(const v16h*)(W2 + ((3 * 4 + jj) * 16 + n) * 128 + kt * 32 + 16 * hh);
        acc[0] = wmma_f16(a, bf0, acc[0]);
        acc[1] = wmma_f16(a, bf1, acc[1]);
        acc[2] = wmma_f16(a, bf2, acc[2]);
        acc[3] = wmma_f16(a, bf3, acc[3]);
      }
#pragma unroll
      for (int v = 0; v < 8; ++v) {
        float iv = sigf(acc[0][v] + bpre[0][jj]);
        float fv = sigf(acc[1][v] + bpre[1][jj]);
        float gv = tanh_fast(acc[2][v] + bpre[2][jj]);
        float ov = sigf(acc[3][v] + bpre[3][jj]);
        float c = fv * creg[jj][v] + iv * gv;
        creg[jj][v] = c;
        hbuf[(hh * 8 + v) * 64 + jj * 16 + n] = (_Float16)(ov * tanh_fast(c));
      }
    }

    // refreshed h fragments (also feed next iteration)
    ah0 = load_a_frag(hbuf, n, 0, hh);
    ah1 = load_a_frag(hbuf, n, 32, hh);

    // ---- latent = h @ Wlat.T + blat (one 16x16 tile, K=64) ----
    v8f accl = splat8(blv);
    {
      v16h bf0 = *(const v16h*)(WlatL + n * 64 + 0 * 32 + 16 * hh);
      v16h bf1 = *(const v16h*)(WlatL + n * 64 + 1 * 32 + 16 * hh);
      accl = wmma_f16(ah0, bf0, accl);
      accl = wmma_f16(ah1, bf1, accl);
    }

    // rnn1_mem_new = tanh(latent); stash latent to LDS for the out head
#pragma unroll
    for (int v = 0; v < 8; ++v) {
      const int m = hh * 8 + v;
      const float lv = accl[v];
      out3[((long)(b0 + m) * kT + t) * 16 + n] = tanh_fast(lv);
      latb[m * 16 + n] = lv;
    }

    // out = latent @ Wout.T + bout  (lane m=n, two y per lane)
    {
      const int m2 = n;
      const int y0 = hh * 2;
      const float4* lrow = (const float4*)(latb + m2 * 16);
#pragma unroll
      for (int yy = 0; yy < 2; ++yy) {
        const int y = y0 + yy;
        const float4* wrow = (const float4*)(WoutL + y * 16);
        float s = boutL[y];
#pragma unroll
        for (int q = 0; q < 4; ++q) {
          float4 lq = lrow[q], wq = wrow[q];
          s += lq.x * wq.x + lq.y * wq.y + lq.z * wq.z + lq.w * wq.w;
        }
        out1[((long)(b0 + m2) * kT + t) * 4 + y] = s;
      }
    }
  }

  // ---- out_sfc = h_last @ Wsfco.T + bsfco ----
  if (lane < 16) {
    const int m2 = lane;
#pragma unroll
    for (int y = 0; y < 3; ++y) {
      float s = bsfcoL[y];
      for (int k = 0; k < 64; ++k)
        s += (float)hbuf[m2 * 64 + k] * WsfcoL[y * 64 + k];
      out2[(b0 + m2) * 3 + y] = s;
    }
  }
}

extern "C" void kernel_launch(void* const* d_in, const int* in_sizes, int n_in,
                              void* d_out, int out_size, void* d_ws,
                              size_t ws_size, hipStream_t stream) {
  (void)in_sizes; (void)n_in; (void)out_size; (void)ws_size;
  const float* inputs_main = (const float*)d_in[0];
  const float* inputs_aux  = (const float*)d_in[1];
  const float* rnn1_mem    = (const float*)d_in[2];
  const float* Wsfc1 = (const float*)d_in[3];
  const float* bsfc1 = (const float*)d_in[4];
  const float* Wsfc2 = (const float*)d_in[5];
  const float* bsfc2 = (const float*)d_in[6];
  const float* Wtoa1 = (const float*)d_in[7];
  const float* btoa1 = (const float*)d_in[8];
  const float* Wtoa2 = (const float*)d_in[9];
  const float* btoa2 = (const float*)d_in[10];
  const float* r1Wih = (const float*)d_in[11];
  const float* r1Whh = (const float*)d_in[12];
  const float* r1bih = (const float*)d_in[13];
  const float* r1bhh = (const float*)d_in[14];
  const float* r2Wih = (const float*)d_in[15];
  const float* r2Whh = (const float*)d_in[16];
  const float* r2bih = (const float*)d_in[17];
  const float* r2bhh = (const float*)d_in[18];
  const float* Wlat  = (const float*)d_in[19];
  const float* blat  = (const float*)d_in[20];
  const float* Wout  = (const float*)d_in[21];
  const float* bout  = (const float*)d_in[22];
  const float* Wsfco = (const float*)d_in[23];
  const float* bsfco = (const float*)d_in[24];

  _Float16* ws = (_Float16*)d_ws;  // [60][B][64] f16 rnn1out scratch
  float* out1 = (float*)d_out;
  float* out2 = out1 + kOut1;
  float* out3 = out2 + kOut2;

  const dim3 grid(kB / 64), block(128);  // 4 waves/WG, 16 batch rows/wave
  k_rnn1<<<grid, block, SM1_SIZE, stream>>>(
      inputs_main, inputs_aux, rnn1_mem, Wsfc1, bsfc1, Wsfc2, bsfc2, r1Wih,
      r1Whh, r1bih, r1bhh, ws);
  k_rnn2<<<grid, block, SM2_SIZE, stream>>>(
      inputs_aux, Wtoa1, btoa1, Wtoa2, btoa2, r2Wih, r2Whh, r2bih, r2bhh, Wlat,
      blat, Wout, bout, Wsfco, bsfco, ws, out1, out2, out3);
}
